// ChannelCompressAttention_36112085024848
// MI455X (gfx1250) — matrix operand order repaired and checked by reading, not compiled, not run on hardware
//
#include <hip/hip_runtime.h>
#include <hip/hip_bf16.h>
#include <math.h>

// Problem constants (B, N, C) = (8, 4096, 1024)
#define BB 8
#define NN 4096
#define CC 1024
#define CHUNKS 64          // row-chunks per batch for streaming passes
#define ROWS 64            // rows per chunk (CHUNKS*ROWS == NN)
#define SCALE 0.03125f     // C^-0.5 == 1/32 exactly

typedef __attribute__((ext_vector_type(8)))  float  v8f;
typedef __attribute__((ext_vector_type(16))) __bf16 v16bf;
typedef __attribute__((ext_vector_type(4)))  float  f4;   // native vec for NT store

// ---------------------------------------------------------------------------
// Deterministic block reduction (256 threads = 8 wave32s). Fixed order.
// ---------------------------------------------------------------------------
__device__ __forceinline__ float block_reduce(float v, float* s_red, float* s_b, bool ismax) {
#pragma unroll
  for (int off = 16; off > 0; off >>= 1) {
    float o = __shfl_xor(v, off, 32);
    v = ismax ? fmaxf(v, o) : (v + o);
  }
  int w = threadIdx.x >> 5;
  if ((threadIdx.x & 31) == 0) s_red[w] = v;
  __syncthreads();
  if (threadIdx.x == 0) {
    float r = s_red[0];
    for (int i = 1; i < 8; ++i) r = ismax ? fmaxf(r, s_red[i]) : (r + s_red[i]);
    *s_b = r;
  }
  __syncthreads();
  return *s_b;   // next call's first barrier protects s_red/s_b reuse
}

// ---------------------------------------------------------------------------
// K0: wq_eff[k] = sum_c w_qkv[c,k] * w_conv1[c]   (q block rows 0..C-1)
// ---------------------------------------------------------------------------
__global__ __launch_bounds__(256) void make_wq_eff(const float* __restrict__ w_qkv,
                                                   const float* __restrict__ w_conv1,
                                                   float* __restrict__ wq) {
  int k = blockIdx.x * 256 + threadIdx.x;
  float acc = 0.f;
  for (int c = 0; c < CC; ++c) acc = fmaf(w_qkv[(size_t)c * CC + k], w_conv1[c], acc);
  wq[k] = acc;
}

// ---------------------------------------------------------------------------
// K1: one streaming pass over x. Per row: agent = x_row . wq_eff (block dot),
// then accumulate y_partial[m] += agent*scale*x_row[m] in registers.
// Writes per-(b,chunk) partials for a deterministic reduce.
// ---------------------------------------------------------------------------
__global__ __launch_bounds__(256) void pass1_agent_y(const float* __restrict__ x,
                                                     const float* __restrict__ wq,
                                                     float* __restrict__ yp) {
  __shared__ float s_w[CC];
  __shared__ float s_red[8];
  __shared__ float s_b;
  const int t = threadIdx.x, b = blockIdx.y, chunk = blockIdx.x;
  for (int i = t; i < CC; i += 256) s_w[i] = wq[i];
  __syncthreads();
  const float w0 = s_w[4*t], w1 = s_w[4*t+1], w2 = s_w[4*t+2], w3 = s_w[4*t+3];
  float a0 = 0.f, a1 = 0.f, a2 = 0.f, a3 = 0.f;
  const float* xb = x + ((size_t)b * NN + (size_t)chunk * ROWS) * CC;
  for (int r = 0; r < ROWS; ++r) {
    if (r + 1 < ROWS) __builtin_prefetch(xb + (size_t)(r + 1) * CC + 4 * t, 0, 1);
    const f4 xv = reinterpret_cast<const f4*>(xb + (size_t)r * CC)[t];
    float p = xv.x * w0 + xv.y * w1 + xv.z * w2 + xv.w * w3;
    float agent = block_reduce(p, s_red, &s_b, false);
    float g = agent * SCALE;
    a0 = fmaf(g, xv.x, a0); a1 = fmaf(g, xv.y, a1);
    a2 = fmaf(g, xv.z, a2); a3 = fmaf(g, xv.w, a3);
  }
  float* dst = yp + (size_t)(b * CHUNKS + chunk) * CC;
  dst[4*t] = a0; dst[4*t+1] = a1; dst[4*t+2] = a2; dst[4*t+3] = a3;
}

// K1b: deterministic reduce of y partials: y[b,m] = sum_chunk yp[b,chunk,m]
__global__ __launch_bounds__(256) void reduce_y(const float* __restrict__ yp,
                                                float* __restrict__ y) {
  int b = blockIdx.y, col = blockIdx.x * 256 + threadIdx.x;
  float acc = 0.f;
  for (int ch = 0; ch < CHUNKS; ++ch)
    acc += yp[(size_t)(b * CHUNKS + ch) * CC + col];
  y[(size_t)b * CC + col] = acc;
}

// ---------------------------------------------------------------------------
// WMMA GEMM: D(8x1024) = A(8x1024) @ op(B)(1024x1024), fp32 in/out.
// A padded to 16 rows (zeros) via branchless row-clamp + multiply-by-mask
// (avoids per-element exec-mask branches in codegen).
// One wave per 16x16 output tile, K-loop of 32.
// fp32 emulated with 3x bf16 WMMA (hi*hi + hi*lo + lo*hi), fp32 accumulate.
// TRANSB=1: B_kn = Bsrc[n*1024 + k]   (i.e. @ W^T with row-major W)
// TRANSB=0: B_kn = Bsrc[k*1024 + n]
// ---------------------------------------------------------------------------
template <int TRANSB>
__global__ __launch_bounds__(32) void gemm8_wmma(const float* __restrict__ A,
                                                 const float* __restrict__ Bsrc,
                                                 float* __restrict__ D) {
  const int lane = threadIdx.x & 31;
  const int mn = lane & 15;        // A: M row; B/D: N column
  const int half = lane >> 4;
  const int n0 = blockIdx.x * 16;
  const int arow = mn & 7;                         // clamped, always valid
  const float amask = (mn < BB) ? 1.0f : 0.0f;     // zero-pad rows 8..15
  v8f acc = {};
  for (int k0 = 0; k0 < CC; k0 += 32) {
    float a_f[16], b_f[16];
#pragma unroll
    for (int j = 0; j < 16; ++j) {
      // 16-bit A 16x32 layout: lanes 0-15 K={0..7,16..23}, lanes 16-31 +8
      int ka = k0 + 8 * half + j + ((j < 8) ? 0 : 8);
      a_f[j] = A[(size_t)arow * CC + ka] * amask;  // branchless
      // 16-bit B 32x16 layout: lanes 0-15 K={0..7,16..23}, lanes 16-31 +8
      int kb = k0 + 8 * half + j + ((j < 8) ? 0 : 8);
      int n = n0 + mn;
      b_f[j] = TRANSB ? Bsrc[(size_t)n * CC + kb] : Bsrc[(size_t)kb * CC + n];
    }
    v16bf ahi, alo, bhi, blo;
#pragma unroll
    for (int j = 0; j < 16; ++j) {
      __bf16 h = (__bf16)a_f[j]; ahi[j] = h; alo[j] = (__bf16)(a_f[j] - (float)h);
      __bf16 g = (__bf16)b_f[j]; bhi[j] = g; blo[j] = (__bf16)(b_f[j] - (float)g);
    }
    acc = __builtin_amdgcn_wmma_f32_16x16x32_bf16(false, ahi, false, bhi, (short)0, acc, false, false);
    acc = __builtin_amdgcn_wmma_f32_16x16x32_bf16(false, ahi, false, blo, (short)0, acc, false, false);
    acc = __builtin_amdgcn_wmma_f32_16x16x32_bf16(false, alo, false, bhi, (short)0, acc, false, false);
  }
  // 32-bit C/D 16x16: VGPR r -> M = r + 8*half, N = lane&15
#pragma unroll
  for (int r = 0; r < 8; ++r) {
    int mrow = r + 8 * half;
    if (mrow < BB) D[(size_t)mrow * CC + n0 + mn] = acc[r];
  }
}

// ---------------------------------------------------------------------------
// K3: in-place softmax over C for each of the 8 rows of attn(8x1024)
// ---------------------------------------------------------------------------
__global__ __launch_bounds__(256) void softmax_c(float* __restrict__ attn) {
  __shared__ float s_red[8];
  __shared__ float s_b;
  const int b = blockIdx.x, t = threadIdx.x;
  float* row = attn + (size_t)b * CC;
  float v[4], mx = -INFINITY;
#pragma unroll
  for (int i = 0; i < 4; ++i) { v[i] = row[4*t + i]; mx = fmaxf(mx, v[i]); }
  mx = block_reduce(mx, s_red, &s_b, true);
  float s = 0.f;
#pragma unroll
  for (int i = 0; i < 4; ++i) { v[i] = __expf(v[i] - mx); s += v[i]; }
  s = block_reduce(s, s_red, &s_b, false);
  float inv = 1.0f / s;
#pragma unroll
  for (int i = 0; i < 4; ++i) row[4*t + i] = v[i] * inv;
}

// ---------------------------------------------------------------------------
// K5: second streaming pass over x (L2-resident: x = 134 MB < 192 MB L2,
// and the only other big traffic -- the output broadcast -- is NT-stored).
// Flash-style online softmax over n fused with the weighted sum
// z[b,m] = sum_n softmax_n(x.wv_eff)[n] * x[n,m].
// Emits per-(b,chunk) unnormalized partials + running (max, denom).
// ---------------------------------------------------------------------------
__global__ __launch_bounds__(256) void pass2_online(const float* __restrict__ x,
                                                    const float* __restrict__ wv,
                                                    float* __restrict__ zp,
                                                    float* __restrict__ zmax,
                                                    float* __restrict__ zden) {
  __shared__ float s_w[CC];
  __shared__ float s_red[8];
  __shared__ float s_b;
  const int t = threadIdx.x, b = blockIdx.y, chunk = blockIdx.x;
  for (int i = t; i < CC; i += 256) s_w[i] = wv[(size_t)b * CC + i];
  __syncthreads();
  const float w0 = s_w[4*t], w1 = s_w[4*t+1], w2 = s_w[4*t+2], w3 = s_w[4*t+3];
  float rm = -INFINITY, rd = 0.f;
  float a0 = 0.f, a1 = 0.f, a2 = 0.f, a3 = 0.f;
  const float* xb = x + ((size_t)b * NN + (size_t)chunk * ROWS) * CC;
  for (int r = 0; r < ROWS; ++r) {
    if (r + 1 < ROWS) __builtin_prefetch(xb + (size_t)(r + 1) * CC + 4 * t, 0, 1);
    const f4 xv = reinterpret_cast<const f4*>(xb + (size_t)r * CC)[t];
    float p = xv.x * w0 + xv.y * w1 + xv.z * w2 + xv.w * w3;
    float L = block_reduce(p, s_red, &s_b, false);
    float mnew = fmaxf(rm, L);
    float f = __expf(rm - mnew);   // first iter: exp(-inf) == 0
    float e = __expf(L - mnew);
    a0 = fmaf(e, xv.x, a0 * f); a1 = fmaf(e, xv.y, a1 * f);
    a2 = fmaf(e, xv.z, a2 * f); a3 = fmaf(e, xv.w, a3 * f);
    rd = fmaf(1.0f, e, rd * f);
    rm = mnew;
  }
  float* dst = zp + (size_t)(b * CHUNKS + chunk) * CC;
  dst[4*t] = a0; dst[4*t+1] = a1; dst[4*t+2] = a2; dst[4*t+3] = a3;
  if (t == 0) { zmax[b * CHUNKS + chunk] = rm; zden[b * CHUNKS + chunk] = rd; }
}

// K5b: deterministic combine of online-softmax partials -> normalized z(8x1024)
__global__ __launch_bounds__(256) void combine_z(const float* __restrict__ zp,
                                                 const float* __restrict__ zmax,
                                                 const float* __restrict__ zden,
                                                 float* __restrict__ z) {
  const int b = blockIdx.y, col = blockIdx.x * 256 + threadIdx.x;
  float M = -INFINITY;
  for (int ch = 0; ch < CHUNKS; ++ch) M = fmaxf(M, zmax[b * CHUNKS + ch]);
  float D = 0.f;
  for (int ch = 0; ch < CHUNKS; ++ch)
    D += __expf(zmax[b * CHUNKS + ch] - M) * zden[b * CHUNKS + ch];
  float acc = 0.f;
  for (int ch = 0; ch < CHUNKS; ++ch)
    acc += __expf(zmax[b * CHUNKS + ch] - M) * zp[(size_t)(b * CHUNKS + ch) * CC + col];
  z[(size_t)b * CC + col] = acc / D;
}

// ---------------------------------------------------------------------------
// K8: out[b,n,c] = out2[b,c] broadcast over n (agent_tok softmax over a
// singleton axis is identically 1). Pure 134 MB streaming store.
// Non-temporal stores (native clang vector type so the builtin accepts it):
// don't let the write-once output evict the L2-resident x tensor.
// ---------------------------------------------------------------------------
__global__ __launch_bounds__(256) void bcast_out(const float* __restrict__ o2,
                                                 float* __restrict__ out) {
  const int t = threadIdx.x, b = blockIdx.y;
  const f4 v = reinterpret_cast<const f4*>(o2 + (size_t)b * CC)[t];
  const int row0 = blockIdx.x * ROWS;
  f4* dst = reinterpret_cast<f4*>(out + ((size_t)b * NN + row0) * CC);
  for (int r = 0; r < ROWS; ++r)
    __builtin_nontemporal_store(v, dst + (size_t)r * (CC / 4) + t);
}

// ---------------------------------------------------------------------------
extern "C" void kernel_launch(void* const* d_in, const int* in_sizes, int n_in,
                              void* d_out, int out_size, void* d_ws, size_t ws_size,
                              hipStream_t stream) {
  const float* x      = (const float*)d_in[0];   // (B,N,C)
  const float* w_qkv  = (const float*)d_in[1];   // (3C,C) row-major
  const float* w_conv = (const float*)d_in[2];   // (C,)
  float* out = (float*)d_out;                    // (B,N,C)

  // Workspace layout (floats). Total ~1.09M floats (~4.4 MB).
  float* ws   = (float*)d_ws;
  size_t off  = 0;
  float* wq   = ws + off; off += CC;               // wq_eff
  float* yp   = ws + off; off += (size_t)BB * CHUNKS * CC; // y partials
  float* y    = ws + off; off += (size_t)BB * CC;  // y = sum_n agent*scale*x
  float* attn = ws + off; off += (size_t)BB * CC;  // logits -> agent_attn
  float* wv   = ws + off; off += (size_t)BB * CC;  // wv_eff
  float* zp   = ws + off; off += (size_t)BB * CHUNKS * CC; // z partials
  float* zmx  = ws + off; off += (size_t)BB * CHUNKS;
  float* zdn  = ws + off; off += (size_t)BB * CHUNKS;
  float* z    = ws + off; off += (size_t)BB * CC;
  float* o2   = ws + off; off += (size_t)BB * CC;  // final per-(b,c) result
  (void)in_sizes; (void)n_in; (void)out_size; (void)ws_size;

  const float* w_k = w_qkv + (size_t)CC * CC;      // rows C..2C-1
  const float* w_v = w_qkv + (size_t)2 * CC * CC;  // rows 2C..3C-1

  make_wq_eff<<<dim3(CC / 256), 256, 0, stream>>>(w_qkv, w_conv, wq);
  pass1_agent_y<<<dim3(CHUNKS, BB), 256, 0, stream>>>(x, wq, yp);
  reduce_y<<<dim3(CC / 256, BB), 256, 0, stream>>>(yp, y);
  gemm8_wmma<1><<<dim3(CC / 16), 32, 0, stream>>>(y, w_k, attn);      // y @ w_k^T
  softmax_c<<<dim3(BB), 256, 0, stream>>>(attn);                      // agent_attn
  gemm8_wmma<0><<<dim3(CC / 16), 32, 0, stream>>>(attn, w_v, wv);     // attn @ w_v
  pass2_online<<<dim3(CHUNKS, BB), 256, 0, stream>>>(x, wv, zp, zmx, zdn);
  combine_z<<<dim3(CC / 256, BB), 256, 0, stream>>>(zp, zmx, zdn, z);
  gemm8_wmma<1><<<dim3(CC / 16), 32, 0, stream>>>(z, w_v, o2);        // z @ w_v^T
  bcast_out<<<dim3(NN / ROWS, BB), 256, 0, stream>>>(o2, out);
}
